// SpatialAttention_25735444037732
// MI455X (gfx1250) — matrix operand order repaired and checked by reading, not compiled, not run on hardware
//
#include <hip/hip_runtime.h>
#include <hip/hip_bf16.h>

typedef __attribute__((ext_vector_type(16))) _Float16 v16h;
typedef __attribute__((ext_vector_type(8)))  _Float16 v8h;
typedef __attribute__((ext_vector_type(8)))  float    v8f;

#define TOK   16384
#define DIMN  1024
#define NH    16
#define HD    64
#define NPOS  256
#define VLD   272        // transposed-V LDS row stride (halves), multiple of 16

static __device__ inline v8f wmma32(v16h a, v16h b, v8f c) {
  return __builtin_amdgcn_wmma_f32_16x16x32_f16(false, a, false, b, (short)0, c, false, false);
}

// A fragment 16x32 (MxK) f16.  Lane L<16: row m=L, K={0..15}; lane L+16:
// row m=L, K={16..31}.  One contiguous 16-half run -> two b128 loads.
static __device__ inline v16h load_a_frag(const _Float16* base, int ld, int lane, int kOff) {
  int m  = lane & 15;
  int k0 = kOff + ((lane >> 4) ? 16 : 0);
  union { v16h v; v8h p[2]; } u;
  u.p[0] = *(const v8h*)(base + m * ld + k0);
  u.p[1] = *(const v8h*)(base + m * ld + k0 + 8);
  return u.v;
}

// B fragment 32x16 (KxN) f16: lane's 16 halves are consecutive along k.
static __device__ inline v16h load_b_frag(const _Float16* ptr) {
  union { v16h v; v8h p[2]; } u;
  u.p[0] = *(const v8h*)(ptr);
  u.p[1] = *(const v8h*)(ptr + 8);
  return u.v;
}

// gfx1250 async copy: 16 bytes global -> LDS, tracked by ASYNCcnt.
// VDST = per-lane LDS byte address (low 32 bits of a generic __shared__ ptr),
// VADDR = per-lane 32-bit byte offset, SADDR = uniform 64-bit base.
static __device__ inline void async_b128_to_lds(void* lds_ptr, uint32_t voff, const void* sbase) {
  uint32_t ldsa = (uint32_t)(uintptr_t)lds_ptr;
  asm volatile("global_load_async_to_lds_b128 %0, %1, %2"
               :
               : "v"(ldsa), "v"(voff), "s"(sbase)
               : "memory");
}

static __device__ inline void wait_async0() {
#if __has_builtin(__builtin_amdgcn_s_wait_asynccnt)
  __builtin_amdgcn_s_wait_asynccnt(0);
#else
  asm volatile("s_wait_asynccnt 0x0" ::: "memory");
#endif
}

// ---------------------------------------------------------------------------
// Kernel 0: one-time f32 -> f16 conversion of x and all four weight matrices.
// ---------------------------------------------------------------------------
__global__ __launch_bounds__(256) void k_cvt(const float* __restrict__ x,
                                             const float* __restrict__ Wq,
                                             const float* __restrict__ Wk,
                                             const float* __restrict__ Wv,
                                             const float* __restrict__ Wo,
                                             _Float16* __restrict__ xh,
                                             _Float16* __restrict__ Wh) {
  size_t idx = (size_t)blockIdx.x * blockDim.x + threadIdx.x;
  const size_t NX = (size_t)TOK * DIMN;
  if (idx < NX) {
    xh[idx] = (_Float16)x[idx];
    return;
  }
  size_t w = idx - NX;                       // 0 .. 4*DIMN*DIMN-1
  int which = (int)(w >> 20);                // DIMN*DIMN = 2^20 per matrix
  size_t off = w & (((size_t)1 << 20) - 1);
  const float* src = (which == 0) ? Wq : (which == 1) ? Wk : (which == 2) ? Wv : Wo;
  Wh[w] = (_Float16)src[off];
}

// ---------------------------------------------------------------------------
// Kernel 1: fused Q/K/V projection  y = x @ W^T  (blockIdx.y selects W)
// ---------------------------------------------------------------------------
__global__ __launch_bounds__(256) void k_qkv_proj(const _Float16* __restrict__ xh,
                                                  const _Float16* __restrict__ Wh,
                                                  float* __restrict__ qraw,
                                                  float* __restrict__ kraw,
                                                  _Float16* __restrict__ vh) {
  const int rb    = blockIdx.x;   // 16-token row block
  const int which = blockIdx.y;   // 0=q, 1=k, 2=v
  const _Float16* W = Wh + ((size_t)which << 20);
  const int tid  = threadIdx.x;
  const int wv   = tid >> 5;
  const int lane = tid & 31;
  const int hi   = lane >> 4;
  const int ln   = lane & 15;

  const int LD = DIMN + 16;
  __shared__ __align__(32) _Float16 xs[16 * (DIMN + 16)];

  // async stage the 16x1024 f16 x tile into LDS
  for (int idx = tid; idx < 16 * DIMN / 8; idx += 256) {
    int r = idx >> 7;              // 128 8-half chunks per row
    int c = (idx & 127) * 8;
    uint32_t voff = (uint32_t)(((rb * 16 + r) * DIMN + c) * 2);
    async_b128_to_lds(xs + r * LD + c, voff, xh);
  }
  wait_async0();
  __syncthreads();

  for (int tt = 0; tt < 8; ++tt) {
    int ct = wv * 8 + tt;          // output column tile, 0..63
    int n  = ct * 16 + ln;         // this lane's B column (W row)
    const _Float16* wp = W + (size_t)n * DIMN + (hi ? 16 : 0);
    v16h b = load_b_frag(wp);      // stage-1 prefetch
    v8f acc = {};
#pragma unroll 4
    for (int kb = 0; kb < DIMN - 32; kb += 32) {
      v16h bn = load_b_frag(wp + kb + 32);      // prefetch next B
      v16h a  = load_a_frag(xs, LD, lane, kb);
      acc = wmma32(a, b, acc);
      b = bn;
    }
    acc = wmma32(load_a_frag(xs, LD, lane, DIMN - 32), b, acc);

    if (which == 2) {
#pragma unroll
      for (int r = 0; r < 8; ++r)
        vh[(size_t)(rb * 16 + r + 8 * hi) * DIMN + ct * 16 + ln] = (_Float16)acc[r];
    } else {
      float* dst = which ? kraw : qraw;
#pragma unroll
      for (int r = 0; r < 8; ++r)
        dst[(size_t)(rb * 16 + r + 8 * hi) * DIMN + ct * 16 + ln] = acc[r];
    }
  }
}

// ---------------------------------------------------------------------------
// Kernel 2: RoPE + qk RMS-norm, f32 -> f16 (one thread per 64-elem head vector)
// ---------------------------------------------------------------------------
__global__ __launch_bounds__(256) void k_rope_norm(const float* __restrict__ qraw,
                                                   const float* __restrict__ kraw,
                                                   const float* __restrict__ cosT,
                                                   const float* __restrict__ sinT,
                                                   _Float16* __restrict__ qh,
                                                   _Float16* __restrict__ kh) {
  int vid   = blockIdx.x * blockDim.x + threadIdx.x;   // 0 .. 2*TOK*NH-1
  int qk    = vid >> 18;                               // TOK*NH = 2^18
  int rest  = vid & ((TOK * NH) - 1);
  int token = rest >> 4;
  int h     = rest & 15;
  int n     = token & (NPOS - 1);                      // sequence position

  const float* src = (qk ? kraw : qraw) + (size_t)token * DIMN + h * HD;
  _Float16*    dst = (qk ? kh   : qh  ) + (size_t)token * DIMN + h * HD;
  const float* cs = cosT + n * HD;
  const float* sn = sinT + n * HD;

  float y[HD];
  float ms = 0.f;
#pragma unroll
  for (int i = 0; i < HD; ++i) {
    float xv  = src[i];
    float rot = (i < 32) ? -src[i + 32] : src[i - 32];
    float o   = xv * cs[i] + rot * sn[i];
    y[i] = o;
    ms  += o * o;
  }
  ms *= (1.0f / HD);
  float rinv = rsqrtf(fmaxf(ms, 1e-6f));
#pragma unroll
  for (int i = 0; i < HD; ++i) dst[i] = (_Float16)(y[i] * rinv);
}

// ---------------------------------------------------------------------------
// Kernel 3: attention per (b,t,h).  512 threads = 16 waves; wave w owns the
// 16-row logit block, softcap+softmax entirely in registers.
// LDS: phase A  q(32K)+k(32K) | vT(34K persistent);  phase B  attn f16 (128K,
// aliases q/k) + vT.  ~162 KB of the 320 KB WGP budget.
// ---------------------------------------------------------------------------
__global__ __launch_bounds__(512) void k_attn(const _Float16* __restrict__ qh,
                                              const _Float16* __restrict__ kh,
                                              const _Float16* __restrict__ vh,
                                              _Float16* __restrict__ ctx) {
  const int bth  = blockIdx.x;
  const int bt   = bth >> 4;
  const int h    = bth & 15;
  const int tid  = threadIdx.x;
  const int wv   = tid >> 5;      // 0..15
  const int lane = tid & 31;
  const int hi   = lane >> 4;
  const int ln   = lane & 15;

  __shared__ __align__(32) _Float16 pool[65536 + HD * VLD];
  _Float16* qs  = pool;                      // [256][64]  (phase A)
  _Float16* ks  = pool + 16384;              // [256][64]  (phase A)
  _Float16* as  = pool;                      // [256][256] (phase B, aliases q/k)
  _Float16* vsT = pool + 65536;              // [64][VLD]  transposed V, persistent

  const size_t base = (size_t)bt * NPOS * DIMN + (size_t)h * HD;
  const _Float16* qb = qh + base;
  const _Float16* kb_ = kh + base;
  for (int idx = tid; idx < NPOS * HD / 8; idx += 512) {
    int r = idx >> 3;              // 8 chunks per 64-wide row
    int c = (idx & 7) * 8;
    uint32_t voff = (uint32_t)((r * DIMN + c) * 2);
    async_b128_to_lds(qs + r * HD + c, voff, qb);
    async_b128_to_lds(ks + r * HD + c, voff, kb_);
  }
  for (int idx = tid; idx < NPOS * HD; idx += 512) {     // transpose V (scalar)
    int r = idx >> 6, c = idx & 63;
    vsT[c * VLD + r] = vh[base + (size_t)r * DIMN + c];
  }
  wait_async0();
  __syncthreads();

  const int wrow = wv * 16;
  v16h aq0 = load_a_frag(qs + wrow * HD, HD, lane, 0);
  v16h aq1 = load_a_frag(qs + wrow * HD, HD, lane, 32);

  // ---- phase A: logits = q @ k^T (B[d][n] = k[n][d], contiguous in d) ----
  v8f acc[16];
#pragma unroll
  for (int j = 0; j < 16; ++j) {
    v8f c = {};
    const _Float16* kr = ks + (j * 16 + ln) * HD + (hi ? 16 : 0);
    c = wmma32(aq0, load_b_frag(kr), c);
    c = wmma32(aq1, load_b_frag(kr + 32), c);
    acc[j] = c;
  }

  // ---- softcap + softmax (row = r + 8*hi; 256 cols = 16 tiles x 16 lanes) --
  const float scale = 0.125f;               // HD^-0.5
#pragma unroll
  for (int j = 0; j < 16; ++j)
#pragma unroll
    for (int r = 0; r < 8; ++r)
      acc[j][r] = 50.0f * tanhf(acc[j][r] * (scale / 50.0f));

#pragma unroll
  for (int r = 0; r < 8; ++r) {
    float mx = -1e30f;
#pragma unroll
    for (int j = 0; j < 16; ++j) mx = fmaxf(mx, acc[j][r]);
    for (int m = 1; m <= 8; m <<= 1) mx = fmaxf(mx, __shfl_xor(mx, m, 32));
    float sum = 0.f;
#pragma unroll
    for (int j = 0; j < 16; ++j) {
      float e = __expf(acc[j][r] - mx);
      acc[j][r] = e;
      sum += e;
    }
    for (int m = 1; m <= 8; m <<= 1) sum += __shfl_xor(sum, m, 32);
    float inv = 1.0f / sum;
#pragma unroll
    for (int j = 0; j < 16; ++j) acc[j][r] *= inv;
  }

  __syncthreads();   // all q/k LDS reads finished before aliasing with attn
#pragma unroll
  for (int j = 0; j < 16; ++j)
#pragma unroll
    for (int r = 0; r < 8; ++r)
      as[(wrow + r + 8 * hi) * NPOS + j * 16 + ln] = (_Float16)acc[j][r];
  __syncthreads();

  // ---- phase B: out = attn @ v ; B[k][n] = vsT[n][k] is contiguous in k ----
#pragma unroll
  for (int ct = 0; ct < 4; ++ct) {
    v8f c = {};
    for (int kb2 = 0; kb2 < NPOS; kb2 += 32) {
      v16h a = load_a_frag(as + wrow * NPOS, NPOS, lane, kb2);
      v16h b = load_b_frag(vsT + (ct * 16 + ln) * VLD + kb2 + (hi ? 16 : 0));
      c = wmma32(a, b, c);
    }
#pragma unroll
    for (int r = 0; r < 8; ++r) {
      int row = wrow + r + 8 * hi;
      ctx[((size_t)bt * NPOS + row) * DIMN + h * HD + ct * 16 + ln] = (_Float16)c[r];
    }
  }
}

// ---------------------------------------------------------------------------
// Kernel 4: output projection  out = ctx @ Wo^T  (f32 result)
// ---------------------------------------------------------------------------
__global__ __launch_bounds__(256) void k_out_proj(const _Float16* __restrict__ ctx,
                                                  const _Float16* __restrict__ Woh,
                                                  float* __restrict__ out) {
  const int rb   = blockIdx.x;
  const int tid  = threadIdx.x;
  const int wv   = tid >> 5;
  const int lane = tid & 31;
  const int hi   = lane >> 4;
  const int ln   = lane & 15;

  const int LD = DIMN + 16;
  __shared__ __align__(32) _Float16 cs[16 * (DIMN + 16)];
  for (int idx = tid; idx < 16 * DIMN / 8; idx += 256) {
    int r = idx >> 7;
    int c = (idx & 127) * 8;
    uint32_t voff = (uint32_t)(((rb * 16 + r) * DIMN + c) * 2);
    async_b128_to_lds(cs + r * LD + c, voff, ctx);
  }
  wait_async0();
  __syncthreads();

  for (int tt = 0; tt < 8; ++tt) {
    int ct = wv * 8 + tt;
    int n  = ct * 16 + ln;
    const _Float16* wp = Woh + (size_t)n * DIMN + (hi ? 16 : 0);
    v16h b = load_b_frag(wp);
    v8f acc = {};
#pragma unroll 4
    for (int kb = 0; kb < DIMN - 32; kb += 32) {
      v16h bn = load_b_frag(wp + kb + 32);
      v16h a  = load_a_frag(cs, LD, lane, kb);
      acc = wmma32(a, b, acc);
      b = bn;
    }
    acc = wmma32(load_a_frag(cs, LD, lane, DIMN - 32), b, acc);
#pragma unroll
    for (int r = 0; r < 8; ++r)
      out[(size_t)(rb * 16 + r + 8 * hi) * DIMN + ct * 16 + ln] = acc[r];
  }
}

// ---------------------------------------------------------------------------
extern "C" void kernel_launch(void* const* d_in, const int* in_sizes, int n_in,
                              void* d_out, int out_size, void* d_ws, size_t ws_size,
                              hipStream_t stream) {
  (void)in_sizes; (void)n_in; (void)out_size; (void)ws_size;
  const float* x  = (const float*)d_in[0];
  const float* rc = (const float*)d_in[1];
  const float* rs = (const float*)d_in[2];
  const float* Wq = (const float*)d_in[3];
  const float* Wk = (const float*)d_in[4];
  const float* Wv = (const float*)d_in[5];
  const float* Wo = (const float*)d_in[6];
  float* out = (float*)d_out;

  char* ws = (char*)d_ws;
  _Float16* xh   = (_Float16*)(ws);                         // 32 MB
  _Float16* Wh   = (_Float16*)(ws + ((size_t)32  << 20));   //  8 MB (Wq,Wk,Wv,Wo)
  float*    qraw = (float*)   (ws + ((size_t)40  << 20));   // 64 MB
  float*    kraw = (float*)   (ws + ((size_t)104 << 20));   // 64 MB
  _Float16* qh   = (_Float16*)(ws + ((size_t)168 << 20));   // 32 MB
  _Float16* kh   = (_Float16*)(ws + ((size_t)200 << 20));   // 32 MB
  _Float16* vh   = (_Float16*)(ws + ((size_t)232 << 20));   // 32 MB
  _Float16* ctx  = (_Float16*)(ws + ((size_t)264 << 20));   // 32 MB
  _Float16* Woh  = Wh + ((size_t)3 << 20);

  const size_t ncvt = (size_t)TOK * DIMN + 4 * (size_t)DIMN * DIMN;
  k_cvt      <<<dim3((unsigned)(ncvt / 256)), 256, 0, stream>>>(x, Wq, Wk, Wv, Wo, xh, Wh);
  k_qkv_proj <<<dim3(TOK / 16, 3), 256, 0, stream>>>(xh, Wh, qraw, kraw, vh);
  k_rope_norm<<<dim3(2 * TOK * NH / 256), 256, 0, stream>>>(qraw, kraw, rc, rs, qh, kh);
  k_attn     <<<dim3((TOK / NPOS) * NH), 512, 0, stream>>>(qh, kh, vh, ctx);
  k_out_proj <<<dim3(TOK / 16), 256, 0, stream>>>(ctx, Woh, out);
}
